// InterpretableMultiHeadAttention_60361470378665
// MI455X (gfx1250) — matrix operand order repaired and checked by reading, not compiled, not run on hardware
//
#include <hip/hip_runtime.h>
#include <hip/hip_bf16.h>

// ---------------- problem constants (fixed by the reference) ----------------
constexpr int B  = 8;
constexpr int S  = 1024;
constexpr int D  = 1024;
constexpr int H  = 16;
constexpr int DK = 64;      // D / H

// ---------------- WMMA types ----------------
typedef __attribute__((ext_vector_type(16))) __bf16 v16bf;
typedef __attribute__((ext_vector_type(8)))  float  v8f;

union Frag {
    v16bf v;
    uint4 q2[2];            // two 16-byte halves
};

// fp32 -> bf16 bits, round-to-nearest-even
__device__ __forceinline__ unsigned short bf_bits(float f) {
    union { float f; unsigned u; } x; x.f = f;
    unsigned r = x.u + 0x7FFFu + ((x.u >> 16) & 1u);
    return (unsigned short)(r >> 16);
}

// ---- fragment loaders over bf16 (row-major, K contiguous) ----
// A-matrix 16-bit 16x32 (ISA 7.12.2): lane holds row m=lane&15;
// lanes 0-15: K base 0, lanes 16-31: K base 8; halves cover K base..base+7
// and base+16..base+23 -> two contiguous 16B loads.
__device__ __forceinline__ v16bf load_a(const unsigned short* row, int k0, int lane) {
    const int base = (lane & 16) ? 8 : 0;
    Frag f;
    f.q2[0] = *reinterpret_cast<const uint4*>(row + k0 + base);
    f.q2[1] = *reinterpret_cast<const uint4*>(row + k0 + base + 16);
    return f.v;
}
// B-matrix 32x16: lane holds column n=lane&15; lanes 0-15: K base 0, lanes 16-31: K base 8;
// halves cover K base..base+7 and base+16..base+23 (two 16B loads).
__device__ __forceinline__ v16bf load_b(const unsigned short* row, int k0, int lane) {
    const int kb = k0 + ((lane & 16) ? 8 : 0);
    Frag f;
    f.q2[0] = *reinterpret_cast<const uint4*>(row + kb);
    f.q2[1] = *reinterpret_cast<const uint4*>(row + kb + 16);
    return f.v;
}

__device__ __forceinline__ v8f wmma_bf16(v16bf a, v16bf b, v8f c) {
    return __builtin_amdgcn_wmma_f32_16x16x32_bf16(false, a, false, b, (short)0, c, false, false);
}

// =====================================================================
// Kernel 0: streaming f32 -> bf16 conversion (one-time pre-pass)
// =====================================================================
__global__ __launch_bounds__(256) void cvt_bf16_kernel(
        const float* __restrict__ src, unsigned short* __restrict__ dst, int n4) {
    const int i = blockIdx.x * blockDim.x + threadIdx.x;
    if (i >= n4) return;
    const float4 f = reinterpret_cast<const float4*>(src)[i];
    union { unsigned short u[4]; uint2 v; } o;
    o.u[0] = bf_bits(f.x); o.u[1] = bf_bits(f.y);
    o.u[2] = bf_bits(f.z); o.u[3] = bf_bits(f.w);
    reinterpret_cast<uint2*>(dst)[i] = o.v;
}

// =====================================================================
// Kernel 1: per-head projection  Y[b,h,s,e] = sum_d X[b,s,d] * W[h,e,d]
// all-bf16 operands; grid = (S/16, B*H), block = 128 (wave w -> e0=16w)
// =====================================================================
__global__ __launch_bounds__(128) void proj_qk_kernel(
        const unsigned short* __restrict__ X, const unsigned short* __restrict__ W,
        unsigned short* __restrict__ Y) {
    const int s0   = blockIdx.x * 16;
    const int bh   = blockIdx.y;
    const int h    = bh % H;
    const int b    = bh / H;
    const int wave = threadIdx.x >> 5;
    const int lane = threadIdx.x & 31;
    const int e0   = wave * 16;
    const int m    = lane & 15;
    const int n    = lane & 15;
    const int msel = (lane & 16) ? 8 : 0;

    const unsigned short* xrow = X + ((size_t)b * S + (s0 + m)) * D;
    const unsigned short* wrow = W + ((size_t)h * DK + (e0 + n)) * D;

    v8f c = {};
#pragma unroll 4
    for (int k0 = 0; k0 < D; k0 += 32)
        c = wmma_bf16(load_a(xrow, k0, lane), load_b(wrow, k0, lane), c);

    unsigned short* yb = Y + (size_t)bh * S * DK;
#pragma unroll
    for (int i = 0; i < 8; ++i)
        yb[(size_t)(s0 + msel + i) * DK + e0 + n] = bf_bits(c[i]);
}

// =====================================================================
// Kernel 2: shared V projection, stored transposed: vsT[b,e,t]
// grid = (S/16, B), block = 128
// =====================================================================
__global__ __launch_bounds__(128) void proj_v_kernel(
        const unsigned short* __restrict__ V, const unsigned short* __restrict__ Wv,
        unsigned short* __restrict__ vsT) {
    const int s0   = blockIdx.x * 16;
    const int b    = blockIdx.y;
    const int wave = threadIdx.x >> 5;
    const int lane = threadIdx.x & 31;
    const int e0   = wave * 16;
    const int m    = lane & 15;
    const int n    = lane & 15;
    const int msel = (lane & 16) ? 8 : 0;

    const unsigned short* xrow = V  + ((size_t)b * S + (s0 + m)) * D;
    const unsigned short* wrow = Wv + (size_t)(e0 + n) * D;

    v8f c = {};
#pragma unroll 4
    for (int k0 = 0; k0 < D; k0 += 32)
        c = wmma_bf16(load_a(xrow, k0, lane), load_b(wrow, k0, lane), c);

#pragma unroll
    for (int i = 0; i < 8; ++i)
        vsT[((size_t)b * DK + (e0 + n)) * S + (s0 + msel + i)] = bf_bits(c[i]);
}

// =====================================================================
// Kernel 3: fused scores + softmax + attention write (NT) + PV
// grid = (S/16, B*H), block = 256 (8 waves)
// LDS: f32 score tile 16x1028 (~64KB) + bf16 prob tile 16x1032 (~32KB)
// =====================================================================
__global__ __launch_bounds__(256) void attn_kernel(
        const unsigned short* __restrict__ qs,
        const unsigned short* __restrict__ ks,
        const unsigned short* __restrict__ vsT,
        float* __restrict__ attn_out,   // [B, Sq, H, Sk]
        float* __restrict__ heads) {    // [B, H, S, DK]
    constexpr int FS = S + 4;           // f32 row stride: 4-bank skew
    constexpr int PS = S + 8;           // bf16 row stride: 16B-aligned rows, 4-bank skew
    __shared__ float          sc[16 * FS];
    __shared__ unsigned short pr[16 * PS];

    const int s0   = blockIdx.x * 16;
    const int bh   = blockIdx.y;
    const int b    = bh / H;
    const int h    = bh % H;
    const int wave = threadIdx.x >> 5;
    const int lane = threadIdx.x & 31;
    const int m    = lane & 15;
    const int n    = lane & 15;
    const int msel = (lane & 16) ? 8 : 0;

    // --- Q fragments (dk=64 -> 2 K-steps), pure b128 loads ---
    const unsigned short* qrow = qs + ((size_t)bh * S + (s0 + m)) * DK;
    const v16bf aq0 = load_a(qrow, 0,  lane);
    const v16bf aq1 = load_a(qrow, 32, lane);

    // --- scores: wave owns k-tiles {wave, wave+8, ...} ---
    for (int t = wave; t < S / 16; t += 8) {
        const unsigned short* krow = ks + ((size_t)bh * S + (t * 16 + n)) * DK;
        v8f c = {};
        c = wmma_bf16(aq0, load_b(krow, 0,  lane), c);
        c = wmma_bf16(aq1, load_b(krow, 32, lane), c);
#pragma unroll
        for (int i = 0; i < 8; ++i)
            sc[(msel + i) * FS + t * 16 + n] = c[i] * 0.125f;  // 1/sqrt(64)
    }
    __syncthreads();

    // --- softmax rows 2w, 2w+1: wave32 shfl reductions ---
#pragma unroll
    for (int rr = 0; rr < 2; ++rr) {
        const int r = wave * 2 + rr;
        const float*    frow = sc + r * FS;
        unsigned short* prow = pr + r * PS;
        float mx = -3.0e38f;
        for (int j = lane; j < S; j += 32) mx = fmaxf(mx, frow[j]);
#pragma unroll
        for (int o = 16; o > 0; o >>= 1) mx = fmaxf(mx, __shfl_xor(mx, o, 32));
        float sum = 0.f;
        float ex[S / 32];
#pragma unroll 8
        for (int jj = 0; jj < S / 32; ++jj) {
            float p = __expf(frow[lane + 32 * jj] - mx);
            ex[jj] = p;
            sum += p;
        }
#pragma unroll
        for (int o = 16; o > 0; o >>= 1) sum += __shfl_xor(sum, o, 32);
        const float inv = 1.0f / sum;
        float* arow = attn_out + (((size_t)b * S + (s0 + r)) * H + h) * S;
#pragma unroll 8
        for (int jj = 0; jj < S / 32; ++jj) {
            const int j = lane + 32 * jj;
            const float p = ex[jj] * inv;
            __builtin_nontemporal_store(p, arow + j);   // write-once, bypass caches
            prow[j] = bf_bits(p);                       // bf16 copy for PV A-fragments
        }
    }
    __syncthreads();

    // --- PV: waves 0..3, N-tile e0 = 16*wave, K = S in steps of 32 ---
    if (wave < 4) {
        const int e0 = wave * 16;
        const unsigned short* vrow = vsT + ((size_t)b * DK + (e0 + n)) * S;
        const unsigned short* arow = pr + m * PS;
        v8f c = {};
#pragma unroll 4
        for (int k0 = 0; k0 < S; k0 += 32)
            c = wmma_bf16(load_a(arow, k0, lane), load_b(vrow, k0, lane), c);
        float* hbase = heads + (size_t)bh * S * DK;
#pragma unroll
        for (int i = 0; i < 8; ++i)
            hbase[(size_t)(s0 + msel + i) * DK + e0 + n] = c[i];
    }
}

// =====================================================================
// Kernel 4: head mean -> bf16  hm[b,s,e] = (1/H) sum_h heads[b,h,s,e]
// =====================================================================
__global__ __launch_bounds__(256) void mean_heads_kernel(
        const float* __restrict__ heads, unsigned short* __restrict__ hm) {
    const size_t total = (size_t)B * S * DK;
    size_t i = (size_t)blockIdx.x * blockDim.x + threadIdx.x;
    if (i >= total) return;
    const size_t e  = i % DK;
    const size_t bs = i / DK;
    const size_t s  = bs % S;
    const size_t b  = bs / S;
    float acc = 0.f;
#pragma unroll
    for (int hh = 0; hh < H; ++hh)
        acc += heads[(((size_t)b * H + hh) * S + s) * DK + e];
    hm[i] = bf_bits(acc * (1.0f / H));
}

// =====================================================================
// Kernel 5: output projection  out[r,d] = sum_e hm[r,e] * Wo[d,e]
// M = B*S = 8192, N = D = 1024, K = 64; all-bf16 operands
// grid = (M/16, N/64), block = 128
// =====================================================================
__global__ __launch_bounds__(128) void out_proj_kernel(
        const unsigned short* __restrict__ hm, const unsigned short* __restrict__ Wo,
        float* __restrict__ out) {
    const int row0 = blockIdx.x * 16;
    const int wave = threadIdx.x >> 5;
    const int lane = threadIdx.x & 31;
    const int n0   = (blockIdx.y * 4 + wave) * 16;
    const int m    = lane & 15;
    const int n    = lane & 15;
    const int msel = (lane & 16) ? 8 : 0;

    const unsigned short* arow = hm + (size_t)(row0 + m) * DK;
    const unsigned short* wrow = Wo + (size_t)(n0 + n) * DK;

    v8f c = {};
#pragma unroll
    for (int k0 = 0; k0 < DK; k0 += 32)
        c = wmma_bf16(load_a(arow, k0, lane), load_b(wrow, k0, lane), c);

#pragma unroll
    for (int i = 0; i < 8; ++i)
        out[(size_t)(row0 + msel + i) * D + n0 + n] = c[i];
}

// =====================================================================
extern "C" void kernel_launch(void* const* d_in, const int* in_sizes, int n_in,
                              void* d_out, int out_size, void* d_ws, size_t ws_size,
                              hipStream_t stream) {
    (void)in_sizes; (void)n_in; (void)out_size; (void)ws_size;

    const float* q  = (const float*)d_in[0];
    const float* k  = (const float*)d_in[1];
    const float* v  = (const float*)d_in[2];
    const float* Wq = (const float*)d_in[3];
    const float* Wk = (const float*)d_in[4];
    const float* Wv = (const float*)d_in[5];
    const float* Wo = (const float*)d_in[6];

    float* out  = (float*)d_out;                      // outputs [B,S,D]
    float* attn = out + (size_t)B * S * D;            // attn    [B,Sq,H,Sk]

    // workspace carve-up (bytes)
    char* ws = (char*)d_ws;
    unsigned short* qb  = (unsigned short*)ws;  ws += (size_t)B * S * D * 2;        // 16 MB
    unsigned short* kb  = (unsigned short*)ws;  ws += (size_t)B * S * D * 2;        // 16 MB
    unsigned short* vb  = (unsigned short*)ws;  ws += (size_t)B * S * D * 2;        // 16 MB
    unsigned short* Wqb = (unsigned short*)ws;  ws += (size_t)H * DK * D * 2;       //  2 MB
    unsigned short* Wkb = (unsigned short*)ws;  ws += (size_t)H * DK * D * 2;       //  2 MB
    unsigned short* Wvb = (unsigned short*)ws;  ws += (size_t)DK * D * 2;           // 128 KB
    unsigned short* Wob = (unsigned short*)ws;  ws += (size_t)D * DK * 2;           // 128 KB
    unsigned short* qs  = (unsigned short*)ws;  ws += (size_t)B * H * S * DK * 2;   // 16 MB
    unsigned short* ks_ = (unsigned short*)ws;  ws += (size_t)B * H * S * DK * 2;   // 16 MB
    unsigned short* vsT = (unsigned short*)ws;  ws += (size_t)B * DK * S * 2;       //  1 MB
    float*          hds = (float*)ws;           ws += (size_t)B * H * S * DK * 4;   // 32 MB
    unsigned short* hm  = (unsigned short*)ws;                                      //  1 MB

    // one-time precision conversion (streaming, HBM-bound, cheap)
    const int nq4 = (B * S * D) / 4, nw4 = (H * DK * D) / 4, nv4 = (DK * D) / 4;
    cvt_bf16_kernel<<<dim3((nq4 + 255) / 256), dim3(256), 0, stream>>>(q,  qb,  nq4);
    cvt_bf16_kernel<<<dim3((nq4 + 255) / 256), dim3(256), 0, stream>>>(k,  kb,  nq4);
    cvt_bf16_kernel<<<dim3((nq4 + 255) / 256), dim3(256), 0, stream>>>(v,  vb,  nq4);
    cvt_bf16_kernel<<<dim3((nw4 + 255) / 256), dim3(256), 0, stream>>>(Wq, Wqb, nw4);
    cvt_bf16_kernel<<<dim3((nw4 + 255) / 256), dim3(256), 0, stream>>>(Wk, Wkb, nw4);
    cvt_bf16_kernel<<<dim3((nv4 + 255) / 256), dim3(256), 0, stream>>>(Wv, Wvb, nv4);
    cvt_bf16_kernel<<<dim3((nv4 + 255) / 256), dim3(256), 0, stream>>>(Wo, Wob, nv4);

    proj_qk_kernel<<<dim3(S / 16, B * H), dim3(128), 0, stream>>>(qb, Wqb, qs);
    proj_qk_kernel<<<dim3(S / 16, B * H), dim3(128), 0, stream>>>(kb, Wkb, ks_);
    proj_v_kernel <<<dim3(S / 16, B),     dim3(128), 0, stream>>>(vb, Wvb, vsT);
    attn_kernel   <<<dim3(S / 16, B * H), dim3(256), 0, stream>>>(qs, ks_, vsT, attn, hds);
    mean_heads_kernel<<<dim3((B * S * DK + 255) / 256), dim3(256), 0, stream>>>(hds, hm);
    out_proj_kernel<<<dim3((B * S) / 16, D / 64), dim3(128), 0, stream>>>(hm, Wob, out);
}
